// MoFo1_4913442587142
// MI455X (gfx1250) — compile-verified, matrix operand order checked by reading, not run on hardware
//
#include <hip/hip_runtime.h>
#include <hip/hip_bf16.h>
#include <stdint.h>

typedef __attribute__((ext_vector_type(16))) __bf16 v16bf;
typedef __attribute__((ext_vector_type(8)))  float  v8f;

#define B_    16
#define T_    720
#define C_    64
#define P_    24
#define PN_   30
#define D_    512
#define H_    8
#define HD_   64
#define L_    2
#define PRED_ 96
#define M_    (B_*C_*P_)   // 24576 rows for token-level GEMMs
#define NROW  (B_*C_)      // 1024 sequences

using u16 = unsigned short;

static __device__ __forceinline__ u16 f2bf(float f) {
  unsigned int u = __float_as_uint(f);
  u += 0x7FFFu + ((u >> 16) & 1u);          // round-to-nearest-even
  return (u16)(u >> 16);
}
static __device__ __forceinline__ float bf2f(u16 h) {
  return __uint_as_float(((unsigned int)h) << 16);
}

union Frag {
  v16bf v;
  uint4 q[2];
};

// ---------------------------------------------------------------------------
// Tiled bf16 WMMA GEMM: C[M,Nc] = A[M,K] * B[K,Nc]  (row-major, f32 accum)
// Block: 256 threads (8 waves). Tile 128x64, K-step 32. Wave -> 32x32 block.
// A tile staged with GLOBAL_LOAD_ASYNC_TO_LDS_B128 (ASYNCcnt); B tile staged
// via VGPRs with an in-register 2-row transpose pack (4x ds_store_b32).
// EPI 0: out bf16 = acc + bias
// EPI 1: out f32  = acc + bias + resid
// EPI 2: dual-B SwiGLU: out bf16 = silu(acc1+bias) * (acc2+bias2)
// EPI 3: out f32  = acc + bias
// GUARD: column bounds checks (only needed for the head GEMM, Nc=96)
// ---------------------------------------------------------------------------
template<int EPI, bool GUARD>
__global__ __launch_bounds__(256)
void gemm_bf16(const u16* __restrict__ A, const u16* __restrict__ Bm,
               const u16* __restrict__ B2,
               const float* __restrict__ bias, const float* __restrict__ bias2,
               const float* __restrict__ resid, void* __restrict__ outp,
               int M, int Nc, int K)
{
  constexpr int LS = 40;                       // LDS row stride (shorts); 16B aligned chunks
  __shared__ u16 sA[128 * LS];
  __shared__ u16 sB[64 * LS];
  __shared__ u16 sB2[(EPI == 2) ? 64 * LS : 8];

  const int tid   = threadIdx.x;
  const int tileM = blockIdx.y * 128;
  const int tileN = blockIdx.x * 64;

  v8f acc[2][2] = {};
  v8f acc2[2][2] = {};

  // A staging map: per-thread 32B = two async b128 chunks
  const int aRow = tid >> 1;                   // 0..127
  const int aSeg = (tid & 1) * 16;             // 0 or 16 shorts
  const unsigned ldsA = (unsigned)(uintptr_t)(&sA[aRow * LS + aSeg]);
  // B staging map: two k-rows x 4 columns, packed to b32 stores
  const int bKK = (tid >> 4) * 2;              // 0,2,..,30
  const int bN0 = (tid & 15) * 4;              // 0..60

  // wave/lane maps
  const int lid  = tid & 31;
  const int wid  = tid >> 5;
  const int wm   = (wid & 3) * 32;             // 4 waves along M
  const int wn   = (wid >> 2) * 32;            // 2 waves along N
  const int lrow = lid & 15;
  const int koff = (lid < 16) ? 0 : 8;         // 16-bit A/B operand K split

  for (int k0 = 0; k0 < K; k0 += 32) {
    { // stage A tile asynchronously into LDS (no VGPR round-trip)
      const u16* g = A + (size_t)(tileM + aRow) * K + k0 + aSeg;
      const unsigned long long ga = (unsigned long long)(uintptr_t)g;
      asm volatile("global_load_async_to_lds_b128 %0, %1, off"
                   :: "v"(ldsA), "v"(ga) : "memory");
      asm volatile("global_load_async_to_lds_b128 %0, %1, off offset:16"
                   :: "v"(ldsA), "v"(ga) : "memory");
    }
    { // stage B tile (transposed into LDS as [n][k])
      unsigned lo0, lo1, hi0, hi1;
      if constexpr (!GUARD) {
        const uint2 r0 = *reinterpret_cast<const uint2*>(Bm + (size_t)(k0 + bKK) * Nc + tileN + bN0);
        const uint2 r1 = *reinterpret_cast<const uint2*>(Bm + (size_t)(k0 + bKK + 1) * Nc + tileN + bN0);
        lo0 = r0.x; lo1 = r0.y; hi0 = r1.x; hi1 = r1.y;
      } else {
        u16 t0[4], t1[4];
        for (int j = 0; j < 4; ++j) {
          const int col = tileN + bN0 + j;
          t0[j] = (col < Nc) ? Bm[(size_t)(k0 + bKK) * Nc + col] : (u16)0;
          t1[j] = (col < Nc) ? Bm[(size_t)(k0 + bKK + 1) * Nc + col] : (u16)0;
        }
        lo0 = t0[0] | ((unsigned)t0[1] << 16); lo1 = t0[2] | ((unsigned)t0[3] << 16);
        hi0 = t1[0] | ((unsigned)t1[1] << 16); hi1 = t1[2] | ((unsigned)t1[3] << 16);
      }
      *reinterpret_cast<unsigned*>(&sB[(bN0 + 0) * LS + bKK]) = (lo0 & 0xFFFFu) | (hi0 << 16);
      *reinterpret_cast<unsigned*>(&sB[(bN0 + 1) * LS + bKK]) = (lo0 >> 16) | (hi0 & 0xFFFF0000u);
      *reinterpret_cast<unsigned*>(&sB[(bN0 + 2) * LS + bKK]) = (lo1 & 0xFFFFu) | (hi1 << 16);
      *reinterpret_cast<unsigned*>(&sB[(bN0 + 3) * LS + bKK]) = (lo1 >> 16) | (hi1 & 0xFFFF0000u);
      if constexpr (EPI == 2) {
        const uint2 r0 = *reinterpret_cast<const uint2*>(B2 + (size_t)(k0 + bKK) * Nc + tileN + bN0);
        const uint2 r1 = *reinterpret_cast<const uint2*>(B2 + (size_t)(k0 + bKK + 1) * Nc + tileN + bN0);
        *reinterpret_cast<unsigned*>(&sB2[(bN0 + 0) * LS + bKK]) = (r0.x & 0xFFFFu) | (r1.x << 16);
        *reinterpret_cast<unsigned*>(&sB2[(bN0 + 1) * LS + bKK]) = (r0.x >> 16) | (r1.x & 0xFFFF0000u);
        *reinterpret_cast<unsigned*>(&sB2[(bN0 + 2) * LS + bKK]) = (r0.y & 0xFFFFu) | (r1.y << 16);
        *reinterpret_cast<unsigned*>(&sB2[(bN0 + 3) * LS + bKK]) = (r0.y >> 16) | (r1.y & 0xFFFF0000u);
      }
    }
    asm volatile("s_wait_asynccnt 0x0" ::: "memory");   // A-tile async copies landed
    __syncthreads();

    Frag a[2], b[2], b2[2];
    for (int i = 0; i < 2; ++i) {
      const int base = (wm + i * 16 + lrow) * LS + koff;
      a[i].q[0] = *reinterpret_cast<const uint4*>(&sA[base]);
      a[i].q[1] = *reinterpret_cast<const uint4*>(&sA[base + 16]);
    }
    for (int j = 0; j < 2; ++j) {
      const int base = (wn + j * 16 + lrow) * LS + koff;
      b[j].q[0] = *reinterpret_cast<const uint4*>(&sB[base]);
      b[j].q[1] = *reinterpret_cast<const uint4*>(&sB[base + 16]);
      if constexpr (EPI == 2) {
        b2[j].q[0] = *reinterpret_cast<const uint4*>(&sB2[base]);
        b2[j].q[1] = *reinterpret_cast<const uint4*>(&sB2[base + 16]);
      }
    }
    for (int i = 0; i < 2; ++i)
      for (int j = 0; j < 2; ++j) {
        acc[i][j] = __builtin_amdgcn_wmma_f32_16x16x32_bf16(
            false, a[i].v, false, b[j].v, (short)0, acc[i][j], false, false);
        if constexpr (EPI == 2)
          acc2[i][j] = __builtin_amdgcn_wmma_f32_16x16x32_bf16(
              false, a[i].v, false, b2[j].v, (short)0, acc2[i][j], false, false);
      }
    __syncthreads();
  }

  // epilogue: C layout — lane<16: N=lane, vgpr r -> M=r; lane>=16: N=lane-16, M=8+r
  for (int i = 0; i < 2; ++i)
    for (int j = 0; j < 2; ++j) {
      const int n = tileN + wn + j * 16 + lrow;
      if constexpr (GUARD) { if (n >= Nc) continue; }
      const int m0 = tileM + wm + i * 16 + ((lid < 16) ? 0 : 8);
      const float bval = bias[n];
      for (int r = 0; r < 8; ++r) {
        const size_t idx = (size_t)(m0 + r) * Nc + n;
        const float val = acc[i][j][r] + bval;
        if constexpr (EPI == 0) {
          ((u16*)outp)[idx] = f2bf(val);
        } else if constexpr (EPI == 1) {
          ((float*)outp)[idx] = val + resid[idx];
        } else if constexpr (EPI == 2) {
          const float v2 = acc2[i][j][r] + bias2[n];
          const float sw = val / (1.0f + __expf(-val));       // silu
          ((u16*)outp)[idx] = f2bf(sw * v2);
        } else {
          ((float*)outp)[idx] = val;
        }
      }
    }
}

// ---------------------------------------------------------------------------
// RevIN statistics: per (b,c), mean/std over T (one wave per row)
// ---------------------------------------------------------------------------
__global__ __launch_bounds__(256)
void meanstd_kernel(const float* __restrict__ x_enc,
                    float* __restrict__ meanB, float* __restrict__ stdB)
{
  const int lid = threadIdx.x & 31, wid = threadIdx.x >> 5;
  const int row = blockIdx.x * 8 + wid;       // b*C + c
  const int b = row >> 6, c = row & 63;
  float s = 0.f, s2 = 0.f;
  for (int t = lid; t < T_; t += 32) {
    const float v = x_enc[((size_t)b * T_ + t) * C_ + c];
    s += v; s2 += v * v;
  }
  for (int m = 16; m; m >>= 1) { s += __shfl_xor(s, m, 32); s2 += __shfl_xor(s2, m, 32); }
  if (lid == 0) {
    const float mean = s * (1.0f / T_);
    const float var  = s2 * (1.0f / T_) - mean * mean;
    meanB[row] = mean;
    stdB[row]  = sqrtf(var + 1e-5f);
  }
}

// ---------------------------------------------------------------------------
// Periodic-sigmoid relative-position log-bias table [L][P][P]
// ---------------------------------------------------------------------------
__global__ __launch_bounds__(128)
void rpbias_kernel(const float* __restrict__ a1, const float* __restrict__ a2,
                   const float* __restrict__ b1, const float* __restrict__ b2,
                   float* __restrict__ biasLog)
{
  const int l = blockIdx.x;
  __shared__ float sa, sb;
  if (threadIdx.x == 0) {
    float da = 0.f, db = 0.f;
    for (int i = 0; i < P_; ++i) {
      da += a1[l * P_ + i] * a2[l * P_ + i];
      db += b1[l * P_ + i] * b2[l * P_ + i];
    }
    sa = 1.0f / (1.0f + __expf(-da));
    sb = (1.0f / (1.0f + __expf(-db))) * (float)P_;
  }
  __syncthreads();
  for (int e = threadIdx.x; e < P_ * P_; e += blockDim.x) {
    const int i = e / P_, j = e % P_;
    const int d0 = (i > j) ? (i - j) : (j - i);
    const float diff = (float)((d0 < P_ - d0) ? d0 : P_ - d0);
    const float val = 1.0f / (1.0f + __expf(sa * (diff - sb)))
                    + __expf(-diff) / (1.0f + __expf(sa * sb));
    biasLog[l * P_ * P_ + e] = logf(val);
  }
}

// ---------------------------------------------------------------------------
// Patch embedding: RevIN-normalize, unfold [P,PN], x@W_in + b_in + cias + bias_p
// One block per (b,c,p)
// ---------------------------------------------------------------------------
__global__ __launch_bounds__(256)
void embed_kernel(const float* __restrict__ x_enc, const float* __restrict__ x_mark,
                  const float* __restrict__ rev_w, const float* __restrict__ rev_b,
                  const float* __restrict__ W_in, const float* __restrict__ b_in,
                  const float* __restrict__ bias_p, const float* __restrict__ cias,
                  const float* __restrict__ meanB, const float* __restrict__ stdB,
                  float* __restrict__ xF)
{
  const int bcp = blockIdx.x;
  const int p  = bcp % P_;
  const int bc = bcp / P_;
  const int c  = bc % C_;
  const int b  = bc / C_;
  __shared__ float xn[PN_];
  __shared__ int cidx;
  if (threadIdx.x < PN_) {
    const int t = p * PN_ + threadIdx.x;
    const float v = x_enc[((size_t)b * T_ + t) * C_ + c];
    xn[threadIdx.x] = (v - meanB[bc]) / stdB[bc] * rev_w[c] + rev_b[c];
  }
  if (threadIdx.x == 32) {
    const float pp = rintf((x_mark[((size_t)b * T_ + (T_ - 1)) * 4 + 0] + 0.5f) * 23.0f);
    int ci = ((int)pp - p) % P_;
    if (ci < 0) ci += P_;
    cidx = ci;
  }
  __syncthreads();
  for (int d = threadIdx.x; d < D_; d += blockDim.x) {
    float acc = b_in[d];
    for (int pn = 0; pn < PN_; ++pn) acc += xn[pn] * W_in[pn * D_ + d];
    acc += cias[cidx * D_ + d] + bias_p[c * D_ + d];
    xF[((size_t)bc * P_ + p) * D_ + d] = acc;
  }
}

// ---------------------------------------------------------------------------
// RMSNorm over D=512 -> bf16 (one wave per row)
// ---------------------------------------------------------------------------
__global__ __launch_bounds__(256)
void rmsnorm_kernel(const float* __restrict__ x, const float* __restrict__ scale,
                    const float* __restrict__ off, u16* __restrict__ out)
{
  const int lid = threadIdx.x & 31, wid = threadIdx.x >> 5;
  const int row = blockIdx.x * 8 + wid;
  const float* xr = x + (size_t)row * D_;
  float vals[16];
  float ss = 0.f;
  for (int i = 0; i < 16; ++i) { vals[i] = xr[lid + 32 * i]; ss += vals[i] * vals[i]; }
  for (int m = 16; m; m >>= 1) ss += __shfl_xor(ss, m, 32);
  const float rms = sqrtf(ss * (1.0f / D_));
  const float inv = 1.0f / (rms + 1e-8f);
  u16* orow = out + (size_t)row * D_;
  for (int i = 0; i < 16; ++i) {
    const int d = lid + 32 * i;
    orow[d] = f2bf(scale[d] * (vals[i] * inv) + off[d]);
  }
}

// ---------------------------------------------------------------------------
// Attention: one wave per (n,h); P=24 tokens, HD=64, log-bias softmax
// qkv layout: [M, 3D] with columns [h][q|k|v][hd]
// ---------------------------------------------------------------------------
__global__ __launch_bounds__(32)
void attn_kernel(const u16* __restrict__ qkv, const float* __restrict__ biasLog,
                 u16* __restrict__ o)
{
  __shared__ float qs[P_ * HD_], ks[P_ * HD_], vs[P_ * HD_];
  const int nh = blockIdx.x;
  const int n = nh >> 3, h = nh & 7;
  const int lid = threadIdx.x;
  const size_t base = (size_t)n * P_ * (3 * D_) + h * (3 * HD_);
  for (int idx = lid; idx < P_ * HD_; idx += 32) {
    const int p = idx >> 6, d = idx & 63;
    const size_t r = base + (size_t)p * (3 * D_);
    qs[idx] = bf2f(qkv[r + d]);
    ks[idx] = bf2f(qkv[r + HD_ + d]);
    vs[idx] = bf2f(qkv[r + 2 * HD_ + d]);
  }
  __syncthreads();
  if (lid < P_) {
    float s[P_];
    float mx = -3.0e38f;
    for (int j = 0; j < P_; ++j) {
      float dot = 0.f;
      for (int d = 0; d < HD_; ++d) dot += qs[lid * HD_ + d] * ks[j * HD_ + d];
      s[j] = dot * 0.125f + biasLog[lid * P_ + j];      // HD^-0.5 = 1/8
      mx = fmaxf(mx, s[j]);
    }
    float sum = 0.f;
    for (int j = 0; j < P_; ++j) { s[j] = __expf(s[j] - mx); sum += s[j]; }
    const float inv = 1.0f / sum;
    u16* orow = o + ((size_t)n * P_ + lid) * D_ + h * HD_;
    for (int d = 0; d < HD_; ++d) {
      float acc = 0.f;
      for (int j = 0; j < P_; ++j) acc += s[j] * vs[j * HD_ + d];
      orow[d] = f2bf(acc * inv);
    }
  }
}

// ---------------------------------------------------------------------------
// f32 -> bf16 conversion (grid-strided)
// ---------------------------------------------------------------------------
__global__ void f32_to_bf16_kernel(const float* __restrict__ src, u16* __restrict__ dst, size_t n)
{
  size_t i = (size_t)blockIdx.x * blockDim.x + threadIdx.x;
  const size_t stride = (size_t)gridDim.x * blockDim.x;
  for (; i < n; i += stride) dst[i] = f2bf(src[i]);
}

// ---------------------------------------------------------------------------
// Final: [B*C, PRED] -> [B, PRED, C] with RevIN denorm
// ---------------------------------------------------------------------------
__global__ __launch_bounds__(256)
void final_kernel(const float* __restrict__ head, const float* __restrict__ rev_w,
                  const float* __restrict__ rev_b, const float* __restrict__ meanB,
                  const float* __restrict__ stdB, float* __restrict__ out)
{
  const int idx = blockIdx.x * blockDim.x + threadIdx.x;   // B*PRED*C = 98304
  const int b = idx / (PRED_ * C_);
  const int r = idx % (PRED_ * C_);
  const int j = r / C_, c = r % C_;
  float v = head[((size_t)b * C_ + c) * PRED_ + j];
  v = (v - rev_b[c]) / (rev_w[c] + 1e-10f);
  v = v * stdB[b * C_ + c] + meanB[b * C_ + c];
  out[idx] = v;
}

// ---------------------------------------------------------------------------
extern "C" void kernel_launch(void* const* d_in, const int* in_sizes, int n_in,
                              void* d_out, int out_size, void* d_ws, size_t ws_size,
                              hipStream_t stream)
{
  (void)in_sizes; (void)n_in; (void)out_size; (void)ws_size;
  const float* x_enc   = (const float*)d_in[0];
  const float* x_mark  = (const float*)d_in[1];
  const float* rev_w   = (const float*)d_in[2];
  const float* rev_b   = (const float*)d_in[3];
  const float* W_in    = (const float*)d_in[4];
  const float* b_in    = (const float*)d_in[5];
  const float* bias_p  = (const float*)d_in[6];
  const float* cias    = (const float*)d_in[7];
  const float* an_scale= (const float*)d_in[8];
  const float* an_off  = (const float*)d_in[9];
  const float* W_qkv   = (const float*)d_in[10];
  const float* b_qkv   = (const float*)d_in[11];
  const float* W_o     = (const float*)d_in[12];
  const float* b_o     = (const float*)d_in[13];
  const float* a1      = (const float*)d_in[14];
  const float* a2      = (const float*)d_in[15];
  const float* b1      = (const float*)d_in[16];
  const float* b2      = (const float*)d_in[17];
  const float* fn_scale= (const float*)d_in[18];
  const float* fn_off  = (const float*)d_in[19];
  const float* W1      = (const float*)d_in[20];
  const float* bW1     = (const float*)d_in[21];
  const float* W2      = (const float*)d_in[22];
  const float* bW2     = (const float*)d_in[23];
  const float* W3      = (const float*)d_in[24];
  const float* bW3     = (const float*)d_in[25];
  const float* W_out   = (const float*)d_in[26];
  const float* b_out   = (const float*)d_in[27];

  // --- workspace carve ---
  char* w = (char*)d_ws;
  auto carve = [&](size_t bytes) -> char* {
    char* p = w; w += (bytes + 255) & ~(size_t)255; return p;
  };
  float* meanB   = (float*)carve((size_t)NROW * 4);
  float* stdB    = (float*)carve((size_t)NROW * 4);
  float* biasLog = (float*)carve((size_t)L_ * P_ * P_ * 4);
  float* xF      = (float*)carve((size_t)M_ * D_ * 4);          // activations f32
  u16*   hB      = (u16*)  carve((size_t)M_ * D_ * 2);          // normed / attn-out bf16
  u16*   bigB    = (u16*)  carve((size_t)M_ * 4 * D_ * 2);      // qkv / FFN-hidden bf16
  float* headF   = (float*)carve((size_t)NROW * PRED_ * 4);
  u16*   WqkvB   = (u16*)  carve((size_t)L_ * D_ * 3 * D_ * 2);
  u16*   WoB     = (u16*)  carve((size_t)L_ * D_ * D_ * 2);
  u16*   W1B     = (u16*)  carve((size_t)L_ * D_ * 4 * D_ * 2);
  u16*   W2B     = (u16*)  carve((size_t)L_ * D_ * 4 * D_ * 2);
  u16*   W3B     = (u16*)  carve((size_t)L_ * 4 * D_ * D_ * 2);
  u16*   WoutB   = (u16*)  carve((size_t)P_ * D_ * PRED_ * 2);

  auto conv = [&](const float* s, u16* d, size_t n) {
    int blocks = (int)((n + 255) / 256);
    if (blocks > 4096) blocks = 4096;
    f32_to_bf16_kernel<<<blocks, 256, 0, stream>>>(s, d, n);
  };

  // weights -> bf16
  conv(W_qkv, WqkvB, (size_t)L_ * D_ * 3 * D_);
  conv(W_o,   WoB,   (size_t)L_ * D_ * D_);
  conv(W1,    W1B,   (size_t)L_ * D_ * 4 * D_);
  conv(W2,    W2B,   (size_t)L_ * D_ * 4 * D_);
  conv(W3,    W3B,   (size_t)L_ * 4 * D_ * D_);
  conv(W_out, WoutB, (size_t)P_ * D_ * PRED_);

  // RevIN stats, RP-bias table, patch embedding
  meanstd_kernel<<<NROW / 8, 256, 0, stream>>>(x_enc, meanB, stdB);
  rpbias_kernel<<<L_, 128, 0, stream>>>(a1, a2, b1, b2, biasLog);
  embed_kernel<<<M_, 256, 0, stream>>>(x_enc, x_mark, rev_w, rev_b, W_in, b_in,
                                       bias_p, cias, meanB, stdB, xF);

  const dim3 blk(256);
  for (int l = 0; l < L_; ++l) {
    // Attention block
    rmsnorm_kernel<<<M_ / 8, blk, 0, stream>>>(xF, an_scale + l * D_, an_off + l * D_, hB);
    gemm_bf16<0, false><<<dim3((3 * D_) / 64, M_ / 128), blk, 0, stream>>>(
        hB, WqkvB + (size_t)l * D_ * 3 * D_, nullptr,
        b_qkv + l * 3 * D_, nullptr, nullptr, (void*)bigB, M_, 3 * D_, D_);
    attn_kernel<<<NROW * H_, 32, 0, stream>>>(bigB, biasLog + l * P_ * P_, hB);
    gemm_bf16<1, false><<<dim3(D_ / 64, M_ / 128), blk, 0, stream>>>(
        hB, WoB + (size_t)l * D_ * D_, nullptr,
        b_o + l * D_, nullptr, xF, (void*)xF, M_, D_, D_);
    // SwiGLU FFN block
    rmsnorm_kernel<<<M_ / 8, blk, 0, stream>>>(xF, fn_scale + l * D_, fn_off + l * D_, hB);
    gemm_bf16<2, false><<<dim3((4 * D_) / 64, M_ / 128), blk, 0, stream>>>(
        hB, W1B + (size_t)l * D_ * 4 * D_, W2B + (size_t)l * D_ * 4 * D_,
        bW1 + l * 4 * D_, bW2 + l * 4 * D_, nullptr, (void*)bigB, M_, 4 * D_, D_);
    gemm_bf16<1, false><<<dim3(D_ / 64, M_ / 128), blk, 0, stream>>>(
        bigB, W3B + (size_t)l * 4 * D_ * D_, nullptr,
        bW3 + l * D_, nullptr, xF, (void*)xF, M_, D_, 4 * D_);
  }

  // Output head: x viewed as [NROW, P*D] (contiguous) @ W_out
  conv(xF, hB, (size_t)M_ * D_);
  gemm_bf16<3, true><<<dim3((PRED_ + 63) / 64, NROW / 128), blk, 0, stream>>>(
      hB, WoutB, nullptr, b_out, nullptr, nullptr, (void*)headF,
      NROW, PRED_, P_ * D_);

  final_kernel<<<(B_ * PRED_ * C_) / 256, 256, 0, stream>>>(
      headF, rev_w, rev_b, meanB, stdB, (float*)d_out);
}
